// ALiBi_5076651344079
// MI455X (gfx1250) — matrix operand order repaired and checked by reading, not compiled, not run on hardware
//
#include <hip/hip_runtime.h>
#include <hip/hip_bf16.h>
#include <math.h>

// CDNA5 WMMA vector types (wave32)
typedef __attribute__((ext_vector_type(16))) __bf16 v16bf;
typedef __attribute__((ext_vector_type(8)))  __bf16 v8bf;
typedef __attribute__((ext_vector_type(8)))  float  v8f;
typedef __attribute__((ext_vector_type(4)))  int    v4i;
typedef __attribute__((ext_vector_type(8)))  int    v8i;

static __device__ __forceinline__ v8f wmma_bf16(v16bf a, v16bf b, v8f c) {
  // (neg_a, A, neg_b, B, c_mod, C, reuse_a, reuse_b)
  return __builtin_amdgcn_wmma_f32_16x16x32_bf16(false, a, false, b, (short)0, c,
                                                 false, false);
}

static __device__ __forceinline__ v16bf cat8(v8bf lo, v8bf hi) {
  return __builtin_shufflevector(lo, hi, 0, 1, 2, 3, 4, 5, 6, 7,
                                 8, 9, 10, 11, 12, 13, 14, 15);
}

// ---------------------------------------------------------------------------
// Tensor Data Mover: 2D tile load Global -> LDS (bf16 elements).
// Builds D# group0 (4 SGPRs) + group1 (8 SGPRs) per CDNA5 ISA
// and issues TENSOR_LOAD_TO_LDS via inline asm (TENSORcnt-tracked).
// ---------------------------------------------------------------------------
static __device__ __forceinline__ void tdm_load_2d_bf16(
    unsigned lds_byte_addr, const void* gaddr,
    unsigned tensor_w,      // tensor_dim0: row length in elements
    unsigned tensor_h,      // tensor_dim1: number of rows
    unsigned row_stride,    // tensor_dim0_stride in elements
    unsigned tile_w,        // tile_dim0 (elements, multiple of 2 dwords)
    unsigned tile_h) {      // tile_dim1 (rows)
  const unsigned long long ga = (unsigned long long)gaddr;
  v4i g0;
  g0[0] = 1;                                        // count=1, user descriptor
  g0[1] = (int)lds_byte_addr;                       // lds_addr [63:32]
  g0[2] = (int)(unsigned)(ga & 0xFFFFFFFFu);        // global_addr[31:0]
  g0[3] = (int)((unsigned)((ga >> 32) & 0x01FFFFFFu) | (2u << 30)); // [56:32], type=2
  v8i g1;
  g1[0] = (int)(1u << 16);                          // data_size=1 -> 2 bytes
  g1[1] = (int)((tensor_w & 0xFFFFu) << 16);        // tensor_dim0 low16 @bits[63:48]
  g1[2] = (int)(((tensor_w >> 16) & 0xFFFFu) |      // tensor_dim0 hi16
                ((tensor_h & 0xFFFFu) << 16));      // tensor_dim1 low16
  g1[3] = (int)(((tensor_h >> 16) & 0xFFFFu) |      // tensor_dim1 hi16
                ((tile_w & 0xFFFFu) << 16));        // tile_dim0
  g1[4] = (int)(tile_h & 0xFFFFu);                  // tile_dim1 (tile_dim2=0)
  g1[5] = (int)row_stride;                          // tensor_dim0_stride[31:0]
  g1[6] = 0;                                        // stride hi16, dim1_stride lo
  g1[7] = 0;
  asm volatile("tensor_load_to_lds %0, %1" ::"s"(g0), "s"(g1) : "memory");
}

// ---------------------------------------------------------------------------
// bf16 fragment loaders (contiguous 16B vector loads)
// A 16x32: lane m = lane&15; K=half*16+e
// B 32x16 (n-major rows): lanes 0-15 K=0..15, lanes 16-31 K=16..31
// ---------------------------------------------------------------------------
static __device__ __forceinline__ v16bf load_a_bf16(const __bf16* base, int ld,
                                                    int lane) {
  const int m    = lane & 15;
  const int half = lane >> 4;
  const __bf16* row = base + (size_t)m * ld;
  v8bf lo = *(const v8bf*)(row + half * 16);
  v8bf hi = *(const v8bf*)(row + half * 16 + 8);
  return cat8(lo, hi);
}

static __device__ __forceinline__ v16bf load_b_bf16(const __bf16* base, int ld,
                                                    int lane) {
  const int n    = lane & 15;
  const int half = lane >> 4;
  const __bf16* row = base + (size_t)n * ld + half * 16;
  v8bf lo = *(const v8bf*)(row);
  v8bf hi = *(const v8bf*)(row + 8);
  return cat8(lo, hi);
}

// ---------------------------------------------------------------------------
// One-shot conversions
// ---------------------------------------------------------------------------
__global__ void __launch_bounds__(256) cvt_f32_bf16(const float* __restrict__ in,
                                                    __bf16* __restrict__ out,
                                                    int n) {
  int i = blockIdx.x * 256 + threadIdx.x;
  if (i < n) out[i] = (__bf16)in[i];
}

// fp32 [K,N] -> bf16 [N,K] transpose via LDS tile
__global__ void __launch_bounds__(256) transpose_cvt_bf16(
    const float* __restrict__ in, __bf16* __restrict__ out, int K, int N) {
  __shared__ float tile[32][33];
  const int n0 = blockIdx.x * 32;
  const int k0 = blockIdx.y * 32;
  const int tx = threadIdx.x & 31;
  const int ty = threadIdx.x >> 5;  // 0..7
#pragma unroll
  for (int r = ty; r < 32; r += 8)
    tile[r][tx] = in[(size_t)(k0 + r) * N + n0 + tx];
  __syncthreads();
#pragma unroll
  for (int r = ty; r < 32; r += 8)
    out[(size_t)(n0 + r) * K + k0 + tx] = (__bf16)tile[tx][r];
}

// ---------------------------------------------------------------------------
// GEMM: out = A[M,K](bf16) @ Wt[N,K]^T (bf16) + bias[N]
// Workgroup = 8 waves, block tile 128x128, K-step 32.
// TDM double-buffers the A panel (128x32) and Wt panel (128x32) into LDS;
// waves read fragments via ds_load_b128 and run 8 WMMAs each per K-step.
// MODE 0: bf16 [M,N]; MODE 1: bf16 transposed [N,M]; MODE 2: f32 [M,N]
// ---------------------------------------------------------------------------
template <int MODE>
__global__ void __launch_bounds__(256) gemm_wmma_tdm(
    const __bf16* __restrict__ A, const __bf16* __restrict__ Wt,
    const float* __restrict__ bias, void* __restrict__ out,
    int M, int N, int K) {
  __shared__ __bf16 sA[2][128 * 32];   // 8 KB per buffer
  __shared__ __bf16 sB[2][128 * 32];

  const int lane  = threadIdx.x & 31;
  const int wid   = threadIdx.x >> 5;          // 0..7
  const int waveM = wid & 3;                   // 4 x 32-row groups
  const int waveN = wid >> 2;                  // 2 x 64-col groups
  const int tilesN = N >> 7;                   // 128-wide N tiles
  const int m0 = (blockIdx.x / tilesN) * 128;
  const int n0 = (blockIdx.x % tilesN) * 128;

  v8f acc[2][4];
#pragma unroll
  for (int r = 0; r < 2; ++r)
#pragma unroll
    for (int c = 0; c < 4; ++c) acc[r][c] = v8f{};

  const unsigned ldsA[2] = {(unsigned)(size_t)&sA[0][0],
                            (unsigned)(size_t)&sA[1][0]};
  const unsigned ldsB[2] = {(unsigned)(size_t)&sB[0][0],
                            (unsigned)(size_t)&sB[1][0]};

  // prologue: stage first K panel
  if (wid == 0) {
    tdm_load_2d_bf16(ldsA[0], A + (size_t)m0 * K, K, M, K, 32, 128);
    tdm_load_2d_bf16(ldsB[0], Wt + (size_t)n0 * K, K, N, K, 32, 128);
  }

  int cur = 0;
  for (int k = 0; k < K; k += 32) {
    if (wid == 0) {
      if (k + 32 < K) {
        tdm_load_2d_bf16(ldsA[cur ^ 1], A + (size_t)m0 * K + (k + 32), K, M, K,
                         32, 128);
        tdm_load_2d_bf16(ldsB[cur ^ 1], Wt + (size_t)n0 * K + (k + 32), K, N, K,
                         32, 128);
        __builtin_amdgcn_s_wait_tensorcnt((short)2);  // current panel landed
      } else {
        __builtin_amdgcn_s_wait_tensorcnt((short)0);
      }
    }
    __syncthreads();

    const __bf16* tA = &sA[cur][0];
    const __bf16* tB = &sB[cur][0];
    v16bf a[2], b[4];
#pragma unroll
    for (int r = 0; r < 2; ++r)
      a[r] = load_a_bf16(tA + (size_t)(waveM * 32 + r * 16) * 32, 32, lane);
#pragma unroll
    for (int c = 0; c < 4; ++c)
      b[c] = load_b_bf16(tB + (size_t)(waveN * 64 + c * 16) * 32, 32, lane);
#pragma unroll
    for (int r = 0; r < 2; ++r)
#pragma unroll
      for (int c = 0; c < 4; ++c)
        acc[r][c] = wmma_bf16(a[r], b[c], acc[r][c]);

    __syncthreads();
    cur ^= 1;
  }

  // C/D layout: lane holds col n = lane&15; VGPR j -> row m = j + 8*(lane>>4)
  const int half  = lane >> 4;
  const int nlane = lane & 15;
#pragma unroll
  for (int r = 0; r < 2; ++r) {
#pragma unroll
    for (int c = 0; c < 4; ++c) {
      const int col = n0 + waveN * 64 + c * 16 + nlane;
      const float bv = bias[col];
#pragma unroll
      for (int j = 0; j < 8; ++j) {
        const int m = m0 + waveM * 32 + r * 16 + j + 8 * half;
        const float v = acc[r][c][j] + bv;
        if constexpr (MODE == 0)
          ((__bf16*)out)[(size_t)m * N + col] = (__bf16)v;
        else if constexpr (MODE == 1)
          ((__bf16*)out)[(size_t)col * M + m] = (__bf16)v;   // contiguous in m
        else
          ((float*)out)[(size_t)m * N + col] = v;
      }
    }
  }
}

// ---------------------------------------------------------------------------
// Flash attention with ALiBi + causal mask.
// One wave per (b, h, 32-query tile); 32 keys per step.
// Q,K: bf16 [B,T,C] row-major; Vt: bf16 [C, B*T]; Y: bf16 [B*T, C].
// ---------------------------------------------------------------------------
__global__ void __launch_bounds__(32) flash_alibi_wmma(
    const __bf16* __restrict__ Q, const __bf16* __restrict__ K,
    const __bf16* __restrict__ Vt, __bf16* __restrict__ Y,
    int T, int C, int nh, int Mtot) {
  const int hd   = C / nh;        // 64
  const int lane = threadIdx.x;
  const int qt   = blockIdx.x;    // 32-row query tile
  const int h    = blockIdx.y;
  const int b    = blockIdx.z;
  const int q0   = qt * 32;

  __shared__ __bf16 sP[32 * 32];  // 32x32 probability tile (bf16)

  const float slope = exp2f(-0.5f * (float)(h + 1));  // nh=16 power-of-two
  const float scale = rsqrtf((float)hd);

  const __bf16* Qb = Q + (size_t)b * T * C + h * hd;
  const __bf16* Kb = K + (size_t)b * T * C + h * hd;

  v16bf qa[2][2];
#pragma unroll
  for (int r = 0; r < 2; ++r)
#pragma unroll
    for (int d = 0; d < 2; ++d)
      qa[r][d] = load_a_bf16(Qb + (size_t)(q0 + r * 16) * C + d * 32, C, lane);

  v8f acc[2][4];
  float mrow[2][8], lrow[2][8];
#pragma unroll
  for (int r = 0; r < 2; ++r) {
#pragma unroll
    for (int c = 0; c < 4; ++c) acc[r][c] = v8f{};
#pragma unroll
    for (int j = 0; j < 8; ++j) { mrow[r][j] = -INFINITY; lrow[r][j] = 0.0f; }
  }

  const int half  = lane >> 4;
  const int nlane = lane & 15;

  for (int k0 = 0; k0 <= q0 + 31; k0 += 32) {
    // ---- S = Q @ K^T : 32x32 tile, head-dim 64 -> 8 WMMAs -----------------
    v16bf kb[2][2];
#pragma unroll
    for (int kc = 0; kc < 2; ++kc)
#pragma unroll
      for (int d = 0; d < 2; ++d)
        kb[kc][d] =
            load_b_bf16(Kb + (size_t)(k0 + kc * 16) * C + d * 32, C, lane);

    v8f s[2][2];
#pragma unroll
    for (int r = 0; r < 2; ++r)
#pragma unroll
      for (int kc = 0; kc < 2; ++kc) {
        v8f t = v8f{};
        t = wmma_bf16(qa[r][0], kb[kc][0], t);
        t = wmma_bf16(qa[r][1], kb[kc][1], t);
        s[r][kc] = t;
      }

    // ---- scale + ALiBi + causal, online softmax ---------------------------
#pragma unroll
    for (int r = 0; r < 2; ++r) {
#pragma unroll
      for (int j = 0; j < 8; ++j) {
        const int qi = q0 + r * 16 + j + 8 * half;
        const int kj0 = k0 + nlane;
        const int kj1 = k0 + 16 + nlane;
        float sv0 = s[r][0][j] * scale;
        float sv1 = s[r][1][j] * scale;
        sv0 = (kj0 <= qi) ? (sv0 - slope * (float)(qi - kj0)) : -INFINITY;
        sv1 = (kj1 <= qi) ? (sv1 - slope * (float)(qi - kj1)) : -INFINITY;

        float rmax = fmaxf(sv0, sv1);
#pragma unroll
        for (int d = 8; d >= 1; d >>= 1)
          rmax = fmaxf(rmax, __shfl_xor(rmax, d, 32));
        const float newm = fmaxf(mrow[r][j], rmax);

        const float p0 = __expf(sv0 - newm);
        const float p1 = __expf(sv1 - newm);
        float rsum = p0 + p1;
#pragma unroll
        for (int d = 8; d >= 1; d >>= 1) rsum += __shfl_xor(rsum, d, 32);

        const float corr = __expf(mrow[r][j] - newm);
        lrow[r][j] = lrow[r][j] * corr + rsum;
        mrow[r][j] = newm;
#pragma unroll
        for (int c = 0; c < 4; ++c) acc[r][c][j] *= corr;

        const int prow = r * 16 + j + 8 * half;
        sP[prow * 32 + nlane]      = (__bf16)p0;
        sP[prow * 32 + 16 + nlane] = (__bf16)p1;
      }
    }
    __syncthreads();

    // ---- reload P rows as A fragments (16B ds loads) ----------------------
    v16bf pa[2];
#pragma unroll
    for (int r = 0; r < 2; ++r) {
      const __bf16* prow = sP + (r * 16 + (lane & 15)) * 32;
      v8bf lo = *(const v8bf*)(prow + half * 16);
      v8bf hi = *(const v8bf*)(prow + half * 16 + 8);
      pa[r] = cat8(lo, hi);
    }

    // ---- acc += P @ V : full K=32, Vt contiguous loads --------------------
#pragma unroll
    for (int c = 0; c < 4; ++c) {
      const __bf16* vrow =
          Vt + (size_t)(h * hd + c * 16 + nlane) * Mtot + (size_t)b * T + k0;
      v8bf lo = *(const v8bf*)(vrow + half * 16);
      v8bf hi = *(const v8bf*)(vrow + half * 16 + 8);
      const v16bf vb = cat8(lo, hi);
#pragma unroll
      for (int r = 0; r < 2; ++r) acc[r][c] = wmma_bf16(pa[r], vb, acc[r][c]);
    }
    __syncthreads();
  }

  // ---- normalize, write Y bf16 [B*T, C] -----------------------------------
#pragma unroll
  for (int r = 0; r < 2; ++r) {
#pragma unroll
    for (int j = 0; j < 8; ++j) {
      const float inv = 1.0f / lrow[r][j];
      const int t = q0 + r * 16 + j + 8 * half;
#pragma unroll
      for (int c = 0; c < 4; ++c) {
        Y[((size_t)b * T + t) * C + h * hd + c * 16 + nlane] =
            (__bf16)(acc[r][c][j] * inv);
      }
    }
  }
}

// ---------------------------------------------------------------------------
// Launch
// ---------------------------------------------------------------------------
extern "C" void kernel_launch(void* const* d_in, const int* in_sizes, int n_in,
                              void* d_out, int out_size, void* d_ws,
                              size_t ws_size, hipStream_t stream) {
  (void)in_sizes; (void)n_in; (void)out_size; (void)ws_size;

  const float* x  = (const float*)d_in[0];
  const float* Wq = (const float*)d_in[1];
  const float* bq = (const float*)d_in[2];
  const float* Wk = (const float*)d_in[3];
  const float* bk = (const float*)d_in[4];
  const float* Wv = (const float*)d_in[5];
  const float* bv = (const float*)d_in[6];
  const float* Wo = (const float*)d_in[7];
  const float* bo = (const float*)d_in[8];

  const int B = 2, T = 2048, C = 1024, NH = 16;
  const int M = B * T;                    // 4096
  const size_t S  = (size_t)M * C;
  const size_t SW = (size_t)C * C;

  __bf16* xb  = (__bf16*)d_ws;
  __bf16* Wqt = xb + S;
  __bf16* Wkt = Wqt + SW;
  __bf16* Wvt = Wkt + SW;
  __bf16* Wot = Wvt + SW;
  __bf16* Qb  = Wot + SW;
  __bf16* Kb  = Qb + S;
  __bf16* Vt  = Kb + S;
  __bf16* Yb  = Vt + S;

  // one-shot conversions
  cvt_f32_bf16<<<dim3((int)(S / 256)), dim3(256), 0, stream>>>(x, xb, (int)S);
  dim3 tgrid(C / 32, C / 32), tblk(256);
  transpose_cvt_bf16<<<tgrid, tblk, 0, stream>>>(Wq, Wqt, C, C);
  transpose_cvt_bf16<<<tgrid, tblk, 0, stream>>>(Wk, Wkt, C, C);
  transpose_cvt_bf16<<<tgrid, tblk, 0, stream>>>(Wv, Wvt, C, C);
  transpose_cvt_bf16<<<tgrid, tblk, 0, stream>>>(Wo, Wot, C, C);

  // projections: 128x128 block tiles (8 waves), TDM-staged LDS panels
  const int blocks = (M / 128) * (C / 128);   // 256
  dim3 gblk(256), ggrid(blocks);
  gemm_wmma_tdm<0><<<ggrid, gblk, 0, stream>>>(xb, Wqt, bq, Qb, M, C, C);
  gemm_wmma_tdm<0><<<ggrid, gblk, 0, stream>>>(xb, Wkt, bk, Kb, M, C, C);
  gemm_wmma_tdm<1><<<ggrid, gblk, 0, stream>>>(xb, Wvt, bv, Vt, M, C, C);

  dim3 agrid(T / 32, NH, B);
  flash_alibi_wmma<<<agrid, dim3(32), 0, stream>>>(Qb, Kb, Vt, Yb, T, C, NH, M);

  gemm_wmma_tdm<2><<<ggrid, gblk, 0, stream>>>(Yb, Wot, bo, d_out, M, C, C);
}
